// MpMaxPoolingMatch_41077067219606
// MI455X (gfx1250) — compile-verified, matrix operand order checked by reading, not run on hardware
//
#include <hip/hip_runtime.h>
#include <hip/hip_bf16.h>

typedef __bf16 bf16;
typedef __attribute__((ext_vector_type(16))) __bf16 bf16x16;
typedef __attribute__((ext_vector_type(8)))  __bf16 bf16x8;
typedef __attribute__((ext_vector_type(4)))  __bf16 bf16x4;
typedef __attribute__((ext_vector_type(8)))  float  f32x8;
typedef __attribute__((ext_vector_type(4)))  float  f32x4;

// Problem constants (from reference): B=32, T=256, D=512, MP=20
constexpr int Bn = 32;
constexpr int Tn = 256;
constexpr int Dn = 512;
constexpr int MPn = 20;

// LDS slab row stride in bf16 elements: 512 + 8 pad -> 1040 B rows -> 4-bank
// rotation per row, conflict-free ds_load_b128 across lanes 0..15.
constexpr int ROW = Dn + 8;

__global__ __launch_bounds__(256)
void MpMaxPoolingMatch_kernel(const float* __restrict__ lt,
                              const float* __restrict__ rt,
                              const float* __restrict__ km,
                              float* __restrict__ out)
{
    __shared__ float sKm[Dn];                 // 2 KB
    __shared__ bf16  slab[2][16 * ROW];       // 2 x 16.25 KB, double buffered

    const int tid  = threadIdx.x;
    const int wave = tid >> 5;
    const int lane = tid & 31;

    const int bid   = blockIdx.x;
    const int b     = bid / (2 * MPn);
    const int rem   = bid % (2 * MPn);
    const int thalf = rem / MPn;
    const int m     = rem % MPn;
    const int t0    = thalf * 128 + wave * 16;   // this wave's 16 t-rows

    // ---- stage km[m,:] into LDS (fp32, used during A construction) ----
    for (int i = tid; i < Dn; i += 256) sKm[i] = km[m * Dn + i];
    __syncthreads();

    // ---- build A fragments: A = lt[b, t0:t0+16, :] * km[m, :]  (bf16) ----
    // 16-bit A 16x32 layout: lanes 0-15 -> M=lane, elems 0..7 = K kb..kb+7,
    // elems 8..15 = K kb+16..kb+23; lanes 16-31 same rows with kb += 8.
    const int arow = t0 + (lane & 15);
    const int koff = (lane >> 4) * 8;
    const float* ltrow = lt + ((size_t)b * Tn + arow) * Dn;

    bf16x16 afrag[16];
#pragma unroll
    for (int ks = 0; ks < 16; ++ks) {
        const int kb = ks * 32 + koff;
        f32x4 x0 = *(const f32x4*)(ltrow + kb);
        f32x4 x1 = *(const f32x4*)(ltrow + kb + 4);
        f32x4 y0 = *(const f32x4*)(ltrow + kb + 16);
        f32x4 y1 = *(const f32x4*)(ltrow + kb + 20);
#pragma unroll
        for (int i = 0; i < 4; ++i) {
            afrag[ks][i]      = (bf16)(x0[i] * sKm[kb + i]);
            afrag[ks][4 + i]  = (bf16)(x1[i] * sKm[kb + 4 + i]);
            afrag[ks][8 + i]  = (bf16)(y0[i] * sKm[kb + 16 + i]);
            afrag[ks][12 + i] = (bf16)(y1[i] * sKm[kb + 20 + i]);
        }
    }

    // ---- cooperative loader: rt[b, s0:s0+16, :] -> bf16 LDS slab ----
    const float* rtb = rt + (size_t)b * Tn * Dn;
    auto load_slab = [&](int buf, int s0) {
        // 16x512 f32 = 2048 float4 loads; 8 per thread, fully coalesced.
#pragma unroll
        for (int j = 0; j < 8; ++j) {
            const int i   = tid + 256 * j;        // float4 index
            const int row = i >> 7;               // 0..15
            const int c4  = (i & 127) << 2;       // col 0..508
            f32x4 v = *(const f32x4*)(rtb + (size_t)(s0 + row) * Dn + c4);
            bf16x4 o;
            o[0] = (bf16)v[0]; o[1] = (bf16)v[1];
            o[2] = (bf16)v[2]; o[3] = (bf16)v[3];
            *(bf16x4*)(&slab[buf][row * ROW + c4]) = o;   // ds_store_b64
        }
    };

    f32x8 rmax;
#pragma unroll
    for (int v = 0; v < 8; ++v) rmax[v] = -3.4e38f;

    load_slab(0, 0);   // prologue

    const int brow = (lane & 15) * ROW;   // B fragment: lane -> N = s-col

    for (int st = 0; st < 16; ++st) {
        __syncthreads();                       // slab[st&1] visible
        if (st + 1 < 16) load_slab((st + 1) & 1, (st + 1) * 16);

        const bf16* slabp = slab[st & 1];
        // Two independent accumulator chains (even/odd k-steps) so adjacent
        // v_wmma instructions have no RAW dependency -> 2-deep WMMA ILP.
        f32x8 c0 = {};
        f32x8 c1 = {};
#pragma unroll
        for (int ks = 0; ks < 16; ks += 2) {
            const int kb0 = ks * 32 + koff;
            const int kb1 = (ks + 1) * 32 + koff;

            bf16x8 lo0 = *(const bf16x8*)(slabp + brow + kb0);
            bf16x8 hi0 = *(const bf16x8*)(slabp + brow + kb0 + 16);
            bf16x8 lo1 = *(const bf16x8*)(slabp + brow + kb1);
            bf16x8 hi1 = *(const bf16x8*)(slabp + brow + kb1 + 16);

            bf16x16 bf0, bf1;
#pragma unroll
            for (int i = 0; i < 8; ++i) {
                bf0[i] = lo0[i]; bf0[8 + i] = hi0[i];
                bf1[i] = lo1[i]; bf1[8 + i] = hi1[i];
            }
            c0 = __builtin_amdgcn_wmma_f32_16x16x32_bf16(
                     false, afrag[ks],     false, bf0, (short)0, c0, false, false);
            c1 = __builtin_amdgcn_wmma_f32_16x16x32_bf16(
                     false, afrag[ks + 1], false, bf1, (short)0, c1, false, false);
        }
        // combine partial sums, then running max over encountered s-columns
#pragma unroll
        for (int v = 0; v < 8; ++v) rmax[v] = fmaxf(rmax[v], c0[v] + c1[v]);
    }

    // ---- cross-lane max within each 16-lane half (over N) ----
#pragma unroll
    for (int off = 8; off >= 1; off >>= 1) {
#pragma unroll
        for (int v = 0; v < 8; ++v) {
            float o = __shfl_xor(rmax[v], off, 32);
            rmax[v] = fmaxf(rmax[v], o);
        }
    }

    // C layout: lanes 0-15 VGPR v -> row M=v ; lanes 16-31 -> row M=v+8
    if ((lane & 15) == 0) {
        const int mbase = (lane >> 4) * 8;
#pragma unroll
        for (int v = 0; v < 8; ++v) {
            const int trow = t0 + mbase + v;
            out[((size_t)b * Tn + trow) * MPn + m] = tanhf(rmax[v]);
        }
    }
}

extern "C" void kernel_launch(void* const* d_in, const int* in_sizes, int n_in,
                              void* d_out, int out_size, void* d_ws, size_t ws_size,
                              hipStream_t stream) {
    const float* lt  = (const float*)d_in[0];   // reps_lt  [B,T,D]
    const float* rt  = (const float*)d_in[1];   // reps_rt  [B,T,D]
    const float* km  = (const float*)d_in[2];   // kernel   [1,1,1,MP,D]
    float*       out = (float*)d_out;           // [B,T,MP]

    dim3 grid(Bn * 2 * MPn);   // 1280 workgroups
    dim3 block(256);           // 8 waves (wave32)
    MpMaxPoolingMatch_kernel<<<grid, block, 0, stream>>>(lt, rt, km, out);
}
